// Attention_23811298689287
// MI455X (gfx1250) — matrix operand order repaired and checked by reading, not compiled, not run on hardware
//
#include <hip/hip_runtime.h>
#include <hip/hip_bf16.h>

#define D_DIM 1024
#define T_DIM 2048
#define B_DIM 32
#define ASTRIDE 1032   // bf16 elems per LDS row: 1024 + 8 pad -> 516 dwords, 516%64=4 -> 16 rows spread over 64 banks

typedef __attribute__((ext_vector_type(16))) __bf16       v16bf;
typedef __attribute__((ext_vector_type(8)))  float        v8f;
typedef __attribute__((ext_vector_type(8)))  unsigned int v8u;

__device__ __forceinline__ unsigned short f2bf(float f) {
    unsigned int u = __builtin_bit_cast(unsigned int, f);
    u += 0x7FFFu + ((u >> 16) & 1u);     // round-to-nearest-even
    return (unsigned short)(u >> 16);
}

// Branch-free tanh: native v_tanh_f32 if the builtin exists, else clamped
// exp-based identity (straight-line VALU/TRANS, no EXEC forking like ocml tanhf).
__device__ __forceinline__ float fast_tanh(float x) {
#if __has_builtin(__builtin_amdgcn_tanhf)
    return __builtin_amdgcn_tanhf(x);
#elif __has_builtin(__builtin_amdgcn_tanh_f32)
    return __builtin_amdgcn_tanh_f32(x);
#else
    // tanh(x) = 1 - 2/(e^{2x}+1); clamp so e^{2x} stays finite, tanh(+-15)==+-1 in fp32
    float xc = fminf(fmaxf(x, -15.0f), 15.0f);
    float e  = __expf(2.0f * xc);
    return 1.0f - 2.0f * __builtin_amdgcn_rcpf(e + 1.0f);
#endif
}

__device__ __forceinline__ v16bf pack_frag(uint4 lo, uint4 hi) {
    v8u u;
    u[0] = lo.x; u[1] = lo.y; u[2] = lo.z; u[3] = lo.w;
    u[4] = hi.x; u[5] = hi.y; u[6] = hi.z; u[7] = hi.w;
    return __builtin_bit_cast(v16bf, u);
}

// Load one K-step's A (LDS) + B (global) fragments: f = {a_lo, a_hi, b_lo, b_hi}
__device__ __forceinline__ void load_kstep(const unsigned short* __restrict__ albase, // LDS row base
                                           const unsigned short* __restrict__ brow,   // global W1^T row
                                           int kk, int hi, uint4 f[4]) {
    f[0] = *(const uint4*)(albase + kk + hi * 8);          // A: K = kk + hi*8 + {0..7}
    f[1] = *(const uint4*)(albase + kk + 16 + hi * 8);     // A: K = kk + 16 + hi*8 + {0..7}
    f[2] = *(const uint4*)(brow + kk + hi * 8);            // B: K = kk + hi*8 + {0..7}
    f[3] = *(const uint4*)(brow + kk + 16 + hi * 8);       // B: K = kk + 16 + hi*8 + {0..7}
}

// ---------------------------------------------------------------------------
// K1: W1 (f32, [k][n]) -> W1^T (bf16, [n][k]) via 32x32 LDS tile transpose
// ---------------------------------------------------------------------------
__global__ __launch_bounds__(256)
void w1_transpose_kernel(const float* __restrict__ w1, unsigned short* __restrict__ w1t) {
    __shared__ float tile[32][33];
    const int k0 = blockIdx.y * 32, n0 = blockIdx.x * 32;
    const int tx = threadIdx.x & 31, ty = threadIdx.x >> 5;   // ty: 0..7
    #pragma unroll
    for (int i = 0; i < 32; i += 8)
        tile[ty + i][tx] = w1[(size_t)(k0 + ty + i) * D_DIM + n0 + tx];
    __syncthreads();
    #pragma unroll
    for (int i = 0; i < 32; i += 8)
        w1t[(size_t)(n0 + ty + i) * D_DIM + k0 + tx] = f2bf(tile[tx][ty + i]);
}

// ---------------------------------------------------------------------------
// K2: sW2[b][n] = sum_k state[b][k] * W2[k][n]
// ---------------------------------------------------------------------------
__global__ __launch_bounds__(256)
void sw2_kernel(const float* __restrict__ s, const float* __restrict__ w2,
                float* __restrict__ out) {
    __shared__ float sh[D_DIM];
    const int b = blockIdx.y;
    const int n = blockIdx.x * 256 + threadIdx.x;
    for (int i = threadIdx.x; i < D_DIM; i += 256) sh[i] = s[b * D_DIM + i];
    __syncthreads();
    float acc = 0.f;
    for (int k = 0; k < D_DIM; ++k)
        acc = fmaf(sh[k], w2[(size_t)k * D_DIM + n], acc);
    out[b * D_DIM + n] = acc;
}

// ---------------------------------------------------------------------------
// K3: fused  scores[b,t] = tanh(enc[b,t,:] @ W1 + sW2[b,:]) . V
// One block = (b, 16 t-rows). 8 waves sweep N in 16-wide slices; K=1024 loop
// of v_wmma_f32_16x16x32_bf16, software-pipelined one K-step (64) ahead.
// ---------------------------------------------------------------------------
__global__ __launch_bounds__(256)
void att_scores_kernel(const float* __restrict__ enc,
                       const unsigned short* __restrict__ w1t,   // bf16 [n][k]
                       const float* __restrict__ sw2,            // [B][D]
                       const float* __restrict__ vvec,           // [D]
                       float* __restrict__ scores) {             // [B][T]
    __shared__ unsigned short Atile[16 * ASTRIDE];
    __shared__ float red[8][16];

    const int b   = blockIdx.y;
    const int t0  = blockIdx.x * 16;
    const int tid = threadIdx.x;

    // ---- stage A: 16 rows x 1024 f32 -> bf16 in LDS (padded rows) ----
    const float4* src = (const float4*)(enc + ((size_t)b * T_DIM + t0) * D_DIM);
    for (int i = tid; i < 16 * 256; i += 256) {
        const int r = i >> 8, c4 = i & 255;
        float4 f = src[r * 256 + c4];
        ushort4 h;
        h.x = f2bf(f.x); h.y = f2bf(f.y); h.z = f2bf(f.z); h.w = f2bf(f.w);
        *(ushort4*)&Atile[r * ASTRIDE + c4 * 4] = h;
    }
    __syncthreads();

    const int lane = tid & 31;
    const int wave = tid >> 5;
    const int nsub = lane & 15;   // = N lane index = A row index
    const int hi   = lane >> 4;   // upper/lower half of the wave

    const unsigned short* albase = &Atile[nsub * ASTRIDE];

    float sacc[8];
    #pragma unroll
    for (int r = 0; r < 8; ++r) sacc[r] = 0.f;

    for (int pass = 0; pass < 8; ++pass) {
        const int n0 = (pass * 8 + wave) * 16;
        const int n  = n0 + nsub;
        const unsigned short* brow = w1t + (size_t)n * D_DIM;

        // hoist epilogue scalars so their latency hides under the GEMM
        const float sw2v = sw2[b * D_DIM + n];
        const float vv   = vvec[n];

        v8f c0 = {0.f, 0.f, 0.f, 0.f, 0.f, 0.f, 0.f, 0.f};
        v8f c1 = {0.f, 0.f, 0.f, 0.f, 0.f, 0.f, 0.f, 0.f};

        // software pipeline: fragments for K-step kk live while kk+64 loads fly
        uint4 f0[4], f1[4], g0[4], g1[4];
        load_kstep(albase, brow, 0, hi, f0);
        load_kstep(albase, brow, 32, hi, f1);

        #pragma unroll 4
        for (int kk = 0; kk < D_DIM; kk += 64) {
            const int kn = (kk + 64) & (D_DIM - 1);   // wrap keeps flow uniform; last prefetch is dead
            load_kstep(albase, brow, kn, hi, g0);
            load_kstep(albase, brow, kn + 32, hi, g1);

            c0 = __builtin_amdgcn_wmma_f32_16x16x32_bf16(false, pack_frag(f0[0], f0[1]),
                                                         false, pack_frag(f0[2], f0[3]),
                                                         (short)0, c0, false, false);
            c1 = __builtin_amdgcn_wmma_f32_16x16x32_bf16(false, pack_frag(f1[0], f1[1]),
                                                         false, pack_frag(f1[2], f1[3]),
                                                         (short)0, c1, false, false);
            #pragma unroll
            for (int j = 0; j < 4; ++j) { f0[j] = g0[j]; f1[j] = g1[j]; }
        }
        v8f c = c0 + c1;

        #pragma unroll
        for (int r = 0; r < 8; ++r)                 // row = r + hi*8
            sacc[r] += vv * fast_tanh(c[r] + sw2v);
    }

    // reduce over the 16 N-lanes inside each half-wave (deterministic butterfly)
    #pragma unroll
    for (int r = 0; r < 8; ++r) {
        float x = sacc[r];
        #pragma unroll
        for (int off = 1; off <= 8; off <<= 1)
            x += __shfl_xor(x, off, 32);
        sacc[r] = x;
    }
    if (nsub == 0) {
        #pragma unroll
        for (int r = 0; r < 8; ++r) red[wave][hi * 8 + r] = sacc[r];
    }
    __syncthreads();
    if (tid < 16) {
        float s = 0.f;
        #pragma unroll
        for (int w = 0; w < 8; ++w) s += red[w][tid];
        scores[(size_t)b * T_DIM + t0 + tid] = s;
    }
}

// ---------------------------------------------------------------------------
// K4: softmax over T per batch
// ---------------------------------------------------------------------------
__global__ __launch_bounds__(256)
void softmax_kernel(const float* __restrict__ scores, float* __restrict__ xe) {
    __shared__ float sh[T_DIM];
    __shared__ float red[8];
    const int b = blockIdx.x, tid = threadIdx.x;

    float lmax = -3.0e38f;
    for (int i = tid; i < T_DIM; i += 256) {
        float v = scores[(size_t)b * T_DIM + i];
        sh[i] = v;
        lmax = fmaxf(lmax, v);
    }
    #pragma unroll
    for (int off = 1; off <= 16; off <<= 1)
        lmax = fmaxf(lmax, __shfl_xor(lmax, off, 32));
    if ((tid & 31) == 0) red[tid >> 5] = lmax;
    __syncthreads();
    float bmax = red[0];
    #pragma unroll
    for (int w = 1; w < 8; ++w) bmax = fmaxf(bmax, red[w]);
    __syncthreads();

    float lsum = 0.f;
    for (int i = tid; i < T_DIM; i += 256) {
        float e = __expf(sh[i] - bmax);
        sh[i] = e;
        lsum += e;
    }
    #pragma unroll
    for (int off = 1; off <= 16; off <<= 1)
        lsum += __shfl_xor(lsum, off, 32);
    if ((tid & 31) == 0) red[tid >> 5] = lsum;
    __syncthreads();
    float bsum = 0.f;
    #pragma unroll
    for (int w = 0; w < 8; ++w) bsum += red[w];
    const float inv = 1.0f / bsum;
    for (int i = tid; i < T_DIM; i += 256)
        xe[(size_t)b * T_DIM + i] = sh[i] * inv;
}

// ---------------------------------------------------------------------------
// K5a: partial con_vec over a 256-wide t chunk (256 blocks -> saturates HBM)
// ---------------------------------------------------------------------------
__global__ __launch_bounds__(256)
void convec_partial_kernel(const float* __restrict__ enc, const float* __restrict__ xe,
                           float* __restrict__ part) {   // [B][8][1024]
    const int b = blockIdx.y, chunk = blockIdx.x, tid = threadIdx.x;
    __shared__ float w[256];
    w[tid] = xe[(size_t)b * T_DIM + chunk * 256 + tid];
    __syncthreads();
    float4 acc = {0.f, 0.f, 0.f, 0.f};
    const float4* p = (const float4*)(enc + ((size_t)b * T_DIM + chunk * 256) * D_DIM);
    for (int t = 0; t < 256; ++t) {
        float4 e = p[(size_t)t * 256 + tid];
        const float s = w[t];
        acc.x = fmaf(s, e.x, acc.x);
        acc.y = fmaf(s, e.y, acc.y);
        acc.z = fmaf(s, e.z, acc.z);
        acc.w = fmaf(s, e.w, acc.w);
    }
    ((float4*)part)[((size_t)b * 8 + chunk) * 256 + tid] = acc;
}

// K5b: deterministic combine of the 8 partials
__global__ __launch_bounds__(256)
void convec_reduce_kernel(const float* __restrict__ part, float* __restrict__ out) {
    const int idx = blockIdx.x * 256 + threadIdx.x;   // b*1024 + d
    const int b = idx >> 10, d = idx & 1023;
    float s = 0.f;
    #pragma unroll
    for (int c = 0; c < 8; ++c) s += part[((size_t)b * 8 + c) * D_DIM + d];
    out[idx] = s;
}

// ---------------------------------------------------------------------------
extern "C" void kernel_launch(void* const* d_in, const int* in_sizes, int n_in,
                              void* d_out, int out_size, void* d_ws, size_t ws_size,
                              hipStream_t stream) {
    const float* state = (const float*)d_in[0];   // [B,D]
    const float* enc   = (const float*)d_in[1];   // [B,T,D]
    const float* W1    = (const float*)d_in[2];   // [D,D]
    const float* W2    = (const float*)d_in[3];   // [D,D]
    const float* V     = (const float*)d_in[4];   // [D,1]

    float* out = (float*)d_out;
    float* con = out;                         // [B,D]   = 32768 floats
    float* xe  = out + B_DIM * D_DIM;         // [B,T,1] = 65536 floats

    char* ws = (char*)d_ws;
    unsigned short* w1t = (unsigned short*)ws;                  // 2 MB bf16
    float* sw2    = (float*)(ws + 2097152);                     // 128 KB
    float* scores = (float*)(ws + 2097152 + 131072);            // 256 KB
    float* part   = (float*)(ws + 2097152 + 131072 + 262144);   // 1 MB

    w1_transpose_kernel<<<dim3(32, 32), 256, 0, stream>>>(W1, w1t);
    sw2_kernel<<<dim3(D_DIM / 256, B_DIM), 256, 0, stream>>>(state, W2, sw2);
    att_scores_kernel<<<dim3(T_DIM / 16, B_DIM), 256, 0, stream>>>(enc, w1t, sw2, V, scores);
    softmax_kernel<<<B_DIM, 256, 0, stream>>>(scores, xe);
    convec_partial_kernel<<<dim3(8, B_DIM), 256, 0, stream>>>(enc, xe, part);
    convec_reduce_kernel<<<(B_DIM * D_DIM) / 256, 256, 0, stream>>>(part, con);
}
